// clique_guided_policy_net_7980049236417
// MI455X (gfx1250) — matrix operand order repaired and checked by reading, not compiled, not run on hardware
//
#include <hip/hip_runtime.h>
#include <hip/hip_bf16.h>
#include <stdint.h>

// ---------------- problem constants (fixed by the harness) ----------------
#define N_   8192
#define K_   15
#define E_   (N_*K_)      // 122880
#define FT_  12
#define DT_  0.25f

typedef __attribute__((ext_vector_type(16))) _Float16 v16h;
typedef __attribute__((ext_vector_type(8)))  _Float16 v8h;
typedef __attribute__((ext_vector_type(8)))  float    v8f;

#define WMMA32(a,b,c) __builtin_amdgcn_wmma_f32_16x16x32_f16(false,(a),false,(b),(short)0,(c),false,false)
#define WAIT_DS() asm volatile("s_wait_dscnt 0x0" ::: "memory")

// ---- A fragment: 16x32 f16 tile from row-major activation, 2x b128 per lane ----
__device__ __forceinline__ v16h ld_a(const _Float16* A, int lda, int row0, int k0, int lane) {
  int m  = lane & 15;
  int kb = (lane < 16) ? 0 : 8;
  const v8h* p = (const v8h*)(A + (size_t)(row0 + m) * lda + k0 + kb);
  v8h lo = p[0];
  v8h hi = p[2];   // +16 halfs
  return __builtin_shufflevector(lo, hi, 0,1,2,3,4,5,6,7,8,9,10,11,12,13,14,15);
}

// ---- B fragment from pre-swizzled tile store (512 halfs per 32x16 tile) ----
__device__ __forceinline__ v16h ld_bp(const _Float16* P, int tile, int lane) {
  const v8h* p = (const v8h*)(P + (size_t)tile * 512 + lane * 16);
  v8h lo = p[0];
  v8h hi = p[1];
  return __builtin_shufflevector(lo, hi, 0,1,2,3,4,5,6,7,8,9,10,11,12,13,14,15);
}

__device__ __forceinline__ float sigf(float x) { return 1.f / (1.f + __expf(-x)); }

// ---------------- utility kernels ----------------
__global__ void k_zero32(uint32_t* p, long n) {
  long i = (long)blockIdx.x * blockDim.x + threadIdx.x;
  if (i < n) p[i] = 0u;
}

// pack a row-major f32 weight [Krows, Ncols] into f16 WMMA B-fragment tiles.
// dst tile (kt0+ktl)*NtilesTotal + (nt0+ntl); element order matches ld_bp.
__global__ void k_pack(const float* __restrict__ src, _Float16* __restrict__ dst,
                       int Krows, int Ncols, int ktCount, int kt0,
                       int NtilesTotal, int nt0, int ntCount) {
  long i = (long)blockIdx.x * blockDim.x + threadIdx.x;
  long total = (long)ktCount * ntCount * 512;
  if (i >= total) return;
  int e    = (int)(i & 15);
  int lane = (int)((i >> 4) & 31);
  int t    = (int)(i >> 9);
  int ktl = t / ntCount, ntl = t % ntCount;
  int kb = (lane < 16) ? 0 : 8;
  int k = ktl * 32 + kb + (e & 7) + ((e >> 3) << 4);   // source row (K)
  int n = ntl * 16 + (lane & 15);            // source col (N)
  float v = (k < Krows && n < Ncols) ? src[(size_t)k * Ncols + n] : 0.f;
  size_t tile = (size_t)(kt0 + ktl) * NtilesTotal + (nt0 + ntl);
  dst[tile * 512 + lane * 16 + e] = (_Float16)v;
}

__global__ void k_biasg(const float* bgh, const float* bgi, float* biasg) {
  int i = blockIdx.x * blockDim.x + threadIdx.x;
  if (i < 1024) biasg[i] = (i < 256) ? bgh[i] : bgi[i - 256];
}

__global__ void k_addv(const float* a, const float* b, float* o, int n) {
  int i = blockIdx.x * blockDim.x + threadIdx.x;
  if (i < n) o[i] = a[i] + b[i];
}

__global__ void k_xz(const float* __restrict__ he, const float* __restrict__ z,
                     _Float16* __restrict__ xzh) {
  long i = (long)blockIdx.x * blockDim.x + threadIdx.x;
  if (i >= (long)N_ * 288) return;
  int n = (int)(i / 288), c = (int)(i % 288);
  float v = (c < 256) ? he[(size_t)n * 256 + c] : ((c < 264) ? z[(size_t)n * 8 + (c - 256)] : 0.f);
  xzh[i] = (_Float16)v;
}

// ---------------- generic WMMA GEMM: C = A(f16) x Bpacked(f16) + bias ----------------
__global__ void k_gemm(const _Float16* __restrict__ A, int lda,
                       const _Float16* __restrict__ Bp,
                       float* __restrict__ C, int ldc,
                       const float* __restrict__ bias,
                       int Mtiles, int Ntiles, int Ktiles) {
  int wave = blockIdx.x * (blockDim.x >> 5) + (threadIdx.x >> 5);
  int lane = threadIdx.x & 31;
  if (wave >= Mtiles * Ntiles) return;
  int mt = wave / Ntiles, nt = wave % Ntiles;
  v8f acc = {};
  for (int kt = 0; kt < Ktiles; ++kt) {
    v16h a = ld_a(A, lda, mt * 16, kt * 32, lane);
    v16h b = ld_bp(Bp, kt * Ntiles + nt, lane);
    acc = WMMA32(a, b, acc);
  }
  int ncol  = nt * 16 + (lane & 15);
  float bv  = bias[ncol];
  int mbase = mt * 16 + ((lane < 16) ? 0 : 8);
#pragma unroll
  for (int r = 0; r < 8; ++r)
    C[(size_t)(mbase + r) * ldc + ncol] = acc[r] + bv;
}

// ---------------- guide GRU ----------------
__global__ void k_ginit(const float* __restrict__ gbuf, float* __restrict__ hg,
                        _Float16* __restrict__ hgh) {
  long i = (long)blockIdx.x * blockDim.x + threadIdx.x;
  if (i >= (long)N_ * 256) return;
  int n = (int)(i >> 8), c = (int)(i & 255);
  float v = gbuf[(size_t)n * 1024 + c];
  hg[i] = v; hgh[i] = (_Float16)v;
}

__global__ void k_gru(const float* __restrict__ gbuf, const float* __restrict__ gh,
                      float* __restrict__ hg, _Float16* __restrict__ hgh) {
  long i = (long)blockIdx.x * blockDim.x + threadIdx.x;
  if (i >= (long)N_ * 256) return;
  int n = (int)(i >> 8), c = (int)(i & 255);
  const float* gb = gbuf + (size_t)n * 1024;
  const float* gr = gh   + (size_t)n * 768;
  float ir = gb[256 + c], iz = gb[512 + c], inn = gb[768 + c];
  float hr = gr[c], hz = gr[256 + c], hn = gr[512 + c];
  float h = hg[i];
  float r = sigf(ir + hr);
  float u = sigf(iz + hz);
  float nh = (1.f - u) * tanhf(inn + r * hn) + u * h;
  hg[i] = nh; hgh[i] = (_Float16)nh;
}

__global__ void k_proj(const float* __restrict__ hg, const float* __restrict__ Wp,
                       const float* __restrict__ bp, float* __restrict__ des_t) {
  int i = blockIdx.x * blockDim.x + threadIdx.x;
  if (i >= N_ * 2) return;
  int n = i >> 1, o = i & 1;
  const float* h = hg + (size_t)n * 256;
  float s = bp[o];
  for (int k = 0; k < 256; ++k) s += h[k] * Wp[k * 2 + o];
  des_t[i] = s;
}

// ---------------- rollout init ----------------
__global__ void k_hc0(const float* __restrict__ xst, const float* __restrict__ Wh0,
                      const float* __restrict__ bh0, const float* __restrict__ Wc0,
                      const float* __restrict__ bc0, _Float16* __restrict__ h0h,
                      float* __restrict__ c0) {
  long i = (long)blockIdx.x * blockDim.x + threadIdx.x;
  if (i >= (long)N_ * 256) return;
  int n = (int)(i >> 8), o = (int)(i & 255);
  const float* x = xst + (size_t)n * 4;
  float h = bh0[o], c = bc0[o];
#pragma unroll
  for (int k = 0; k < 4; ++k) { h += x[k] * Wh0[k * 256 + o]; c += x[k] * Wc0[k * 256 + o]; }
  h0h[i] = (_Float16)h; c0[i] = c;
}

__global__ void k_nodes0(const float* __restrict__ x_last, const float* __restrict__ x_st_last,
                         const float* __restrict__ z, float* __restrict__ prev,
                         float* __restrict__ prev_st, _Float16* __restrict__ a_in) {
  int n = blockIdx.x * blockDim.x + threadIdx.x;
  if (n >= N_) return;
#pragma unroll
  for (int j = 0; j < 4; ++j) { prev[n*4+j] = x_last[n*4+j]; prev_st[n*4+j] = x_st_last[n*4+j]; }
  v8h zz;
#pragma unroll
  for (int j = 0; j < 8; ++j) zz[j] = (_Float16)z[n*8+j];
  *(v8h*)(a_in + (size_t)n * 288 + 256) = zz;
}

__global__ void k_rad(const float* __restrict__ node_size, const int* __restrict__ e1,
                      const int* __restrict__ e2, float* __restrict__ rad) {
  int e = blockIdx.x * blockDim.x + threadIdx.x;
  if (e >= E_) return;
  int i1 = e1[e], i2 = e2[e];
  float a0 = node_size[i1*2], a1 = node_size[i1*2+1];
  float b0 = node_size[i2*2], b1 = node_size[i2*2+1];
  rad[e] = 0.5f * (sqrtf(a0*a0 + a1*a1) + sqrtf(b0*b0 + b1*b1));
}

// ---------------- per-step: edge features + collision cost ----------------
__global__ void k_edge(const float* __restrict__ prev, const float* __restrict__ node_size,
                       const int* __restrict__ e1, const int* __restrict__ e2,
                       const float* __restrict__ rad, _Float16* __restrict__ feat,
                       float* __restrict__ ccost) {
  int e = blockIdx.x * blockDim.x + threadIdx.x;
  float val = 0.f;
  if (e < E_) {
    int i1 = e1[e], i2 = e2[e];
    const float* s1 = prev + (size_t)i1 * 4;
    const float* s2 = prev + (size_t)i2 * 4;
    float dx = s1[0] - s2[0], dy = s1[1] - s2[1];
    float dist = sqrtf(dx*dx + dy*dy);
    float rc = rad[e] - dist;
    val = rc > 0.f ? rc : 0.f;
    v8h f;
    f[0] = (_Float16)(s2[0]-s1[0]); f[1] = (_Float16)(s2[1]-s1[1]);
    f[2] = (_Float16)(s2[2]-s1[2]); f[3] = (_Float16)(s2[3]-s1[3]);
    f[4] = (_Float16)node_size[i1*2+0]; f[5] = (_Float16)node_size[i1*2+1];
    f[6] = (_Float16)node_size[i2*2+0]; f[7] = (_Float16)node_size[i2*2+1];
    *(v8h*)(feat + (size_t)e * 32) = f;
  }
#pragma unroll
  for (int m = 16; m; m >>= 1) val += __shfl_xor(val, m, 32);
  if ((threadIdx.x & 31) == 0 && val != 0.f) atomicAdd(ccost, val);
}

// ---------------- per-step: fused 2-layer edge encoder (WMMA) ----------------
__global__ void k_enc(const _Float16* __restrict__ feat, const _Float16* __restrict__ We1p,
                      const float* __restrict__ be1, const _Float16* __restrict__ We2p,
                      const float* __restrict__ be2, _Float16* __restrict__ enc_all) {
  __shared__ __align__(16) _Float16 hid[4 * 16 * 256];
  int wib = threadIdx.x >> 5, lane = threadIdx.x & 31;
  int wave = blockIdx.x * 4 + wib;
  if (wave >= E_ / 16) return;
  _Float16* H = hid + wib * 4096;
  int row0 = wave * 16;
  int nloc = lane & 15, mbase = (lane < 16) ? 0 : 8;

  v16h a = ld_a(feat, 32, row0, 0, lane);        // K padded 8->32 with zeros
#pragma unroll
  for (int nt = 0; nt < 16; ++nt) {
    v16h b = ld_bp(We1p, nt, lane);
    v8f  c = {};
    c = WMMA32(a, b, c);
    float bv = be1[nt * 16 + nloc];
#pragma unroll
    for (int r = 0; r < 8; ++r) {
      float v = c[r] + bv; v = v > 0.f ? v : 0.f;
      H[(mbase + r) * 256 + nt * 16 + nloc] = (_Float16)v;
    }
  }
  WAIT_DS();
#pragma unroll
  for (int nt = 0; nt < 8; ++nt) {
    v8f c = {};
#pragma unroll
    for (int kt = 0; kt < 8; ++kt) {
      v16h a2 = ld_a(H, 256, 0, kt * 32, lane);
      v16h b2 = ld_bp(We2p, kt * 8 + nt, lane);
      c = WMMA32(a2, b2, c);
    }
    float bv = be2[nt * 16 + nloc];
    _Float16* dst = enc_all + (size_t)(row0 + 1) * 128;   // +1: pad row 0
#pragma unroll
    for (int r = 0; r < 8; ++r)
      dst[(size_t)(mbase + r) * 128 + nt * 16 + nloc] = (_Float16)(c[r] + bv);
  }
}

// ---------------- per-step: attention + context (one wave per node) ----------------
__global__ void k_att(int t, const float* __restrict__ prev_st, const float* __restrict__ x_st_last,
                      const float* __restrict__ des, const _Float16* __restrict__ enc_all,
                      const int* __restrict__ obs_idx, const _Float16* __restrict__ Watt1p,
                      const float* __restrict__ Watt2, const float* __restrict__ vatt,
                      _Float16* __restrict__ ctxh, _Float16* __restrict__ a_in,
                      float* __restrict__ terr_sq) {
  __shared__ __align__(16) _Float16 obs_s[4 * 16 * 128];
  __shared__ float sc_s[4 * 16];
  int wib = threadIdx.x >> 5, lane = threadIdx.x & 31;
  int n = blockIdx.x * 4 + wib;
  if (n >= N_) return;
  _Float16* obs = obs_s + wib * 2048;
  float* sc = sc_s + wib * 16;

  const int* oi = obs_idx + (size_t)n * K_;
  // gather 15 rows of 128 halfs (16B vectors), row 15 zeroed
  for (int idx = lane; idx < 256; idx += 32) {
    int j = idx >> 4, q = idx & 15;
    uint4 v = make_uint4(0u, 0u, 0u, 0u);
    if (j < K_) {
      int row = oi[j];
      v = ((const uint4*)(enc_all + (size_t)row * 128))[q];
    }
    ((uint4*)obs)[idx] = v;
  }

  float m = (t == 0) ? 0.f : 1.f;
  const float* ps = prev_st + (size_t)n * 4;
  const float* xs = x_st_last + (size_t)n * 4;
  float dpx = (t > 0) ? des[(size_t)(t-1)*N_*2 + n*2 + 0] : 0.f;
  float dpy = (t > 0) ? des[(size_t)(t-1)*N_*2 + n*2 + 1] : 0.f;
  float rel0 = ps[0] - m*xs[0] - m*dpx;
  float rel1 = ps[1] - m*xs[1] - m*dpy;
  float rel2 = ps[2] - m*xs[2];
  float rel3 = ps[3] - m*xs[3];
  float dtx = des[(size_t)t*N_*2 + n*2], dty = des[(size_t)t*N_*2 + n*2 + 1];
  int t2 = (t + 1 < FT_) ? (t + 1) : (FT_ - 1);
  float dnx = des[(size_t)t2*N_*2 + n*2], dny = des[(size_t)t2*N_*2 + n*2 + 1];
  float wp0 = dtx - ps[0], wp1 = dty - ps[1];
  float wq0 = dnx - ps[0], wq1 = dny - ps[1];
  if (lane == 0) {
    atomicAdd(terr_sq + t, rel0*rel0 + rel1*rel1);
    v8h ai;
    ai[0]=(_Float16)rel0; ai[1]=(_Float16)rel1; ai[2]=(_Float16)rel2; ai[3]=(_Float16)rel3;
    ai[4]=(_Float16)wp0;  ai[5]=(_Float16)wp1;  ai[6]=(_Float16)wq0;  ai[7]=(_Float16)wq1;
    *(v8h*)(a_in + (size_t)n * 288 + 264) = ai;
  }
  WAIT_DS();

  v8f c = {};
#pragma unroll
  for (int kt = 0; kt < 4; ++kt) {
    v16h a = ld_a(obs, 128, 0, kt * 32, lane);
    v16h b = ld_bp(Watt1p, kt, lane);
    c = WMMA32(a, b, c);
  }
  int d = lane & 15, mb = (lane < 16) ? 0 : 8;
  float ra = rel0*Watt2[d] + rel1*Watt2[16+d] + rel2*Watt2[32+d] + rel3*Watt2[48+d];
  float va = vatt[d];
  float s8[8];
#pragma unroll
  for (int r = 0; r < 8; ++r) s8[r] = tanhf(c[r] + ra) * va;
#pragma unroll
  for (int msk = 1; msk < 16; msk <<= 1)
#pragma unroll
    for (int r = 0; r < 8; ++r) s8[r] += __shfl_xor(s8[r], msk, 32);
  if ((lane & 15) == 0)
#pragma unroll
    for (int r = 0; r < 8; ++r) sc[mb + r] = s8[r];
  WAIT_DS();

  float mx = -1e30f;
  for (int j = 0; j < K_; ++j) mx = fmaxf(mx, sc[j]);
  float att[K_]; float ssum = 0.f;
  for (int j = 0; j < K_; ++j) { att[j] = __expf(sc[j] - mx); ssum += att[j]; }
  float inv = 1.f / ssum;
#pragma unroll
  for (int q = 0; q < 4; ++q) {
    int cc = lane * 4 + q;
    float s = 0.f;
    for (int j = 0; j < K_; ++j) s += att[j] * (float)obs[j * 128 + cc];
    ctxh[(size_t)n * 128 + cc] = (_Float16)(s * inv);
  }
}

// ---------------- per-step: fused LSTM GEMM (K=384) + cell update ----------------
__global__ void k_lstm(const _Float16* __restrict__ ctxh, const _Float16* __restrict__ hin,
                       const _Float16* __restrict__ Wlp, const float* __restrict__ bl,
                       float* __restrict__ cst, _Float16* __restrict__ hout,
                       _Float16* __restrict__ a_in) {
  int wave = blockIdx.x * (blockDim.x >> 5) + (threadIdx.x >> 5);
  int lane = threadIdx.x & 31;
  if (wave >= 512 * 16) return;
  int mt = wave >> 4, ct = wave & 15;
  v8f gi = {}, gf = {}, gg = {}, go = {};
  for (int kt = 0; kt < 12; ++kt) {
    v16h a = (kt < 4) ? ld_a(ctxh, 128, mt * 16, kt * 32, lane)
                      : ld_a(hin, 256, mt * 16, (kt - 4) * 32, lane);
    v16h b0 = ld_bp(Wlp, kt * 64 + ct, lane);       gi = WMMA32(a, b0, gi);
    v16h b1 = ld_bp(Wlp, kt * 64 + 16 + ct, lane);  gf = WMMA32(a, b1, gf);
    v16h b2 = ld_bp(Wlp, kt * 64 + 32 + ct, lane);  gg = WMMA32(a, b2, gg);
    v16h b3 = ld_bp(Wlp, kt * 64 + 48 + ct, lane);  go = WMMA32(a, b3, go);
  }
  int col = ct * 16 + (lane & 15);
  float bi = bl[col], bf = bl[256 + col], bg = bl[512 + col], bo = bl[768 + col];
  int mbase = mt * 16 + ((lane < 16) ? 0 : 8);
#pragma unroll
  for (int r = 0; r < 8; ++r) {
    size_t idx = (size_t)(mbase + r) * 256 + col;
    float cc = cst[idx];
    cc = sigf(gf[r] + bf) * cc + sigf(gi[r] + bi) * tanhf(gg[r] + bg);
    float hh = sigf(go[r] + bo) * tanhf(cc);
    cst[idx] = cc;
    _Float16 hhh = (_Float16)hh;
    hout[idx] = hhh;
    a_in[(size_t)(mbase + r) * 288 + col] = hhh;
  }
}

// ---------------- per-step: action MLP (WMMA + VALU head) + dynamics ----------------
__global__ void k_act(int t, const _Float16* __restrict__ a_in, const _Float16* __restrict__ Wa1p,
                      const float* __restrict__ ba1, const float* __restrict__ Wa2,
                      const float* __restrict__ ba2, float* __restrict__ prev,
                      float* __restrict__ prev_st, float* __restrict__ out_state,
                      float* __restrict__ out_input) {
  __shared__ __align__(16) _Float16 hid[4 * 16 * 256];
  __shared__ float u_s[4 * 32];
  __shared__ float wa2_s[512];
  for (int q = threadIdx.x; q < 512; q += blockDim.x) wa2_s[q] = Wa2[q];
  __syncthreads();
  int wib = threadIdx.x >> 5, lane = threadIdx.x & 31;
  int wave = blockIdx.x * 4 + wib;
  if (wave >= 512) return;
  _Float16* H = hid + wib * 4096;
  int row0 = wave * 16;
  int nloc = lane & 15, mbase = (lane < 16) ? 0 : 8;
#pragma unroll
  for (int nt = 0; nt < 16; ++nt) {
    v8f c = {};
    for (int kt = 0; kt < 9; ++kt) {
      v16h a = ld_a(a_in, 288, row0, kt * 32, lane);
      v16h b = ld_bp(Wa1p, kt * 16 + nt, lane);
      c = WMMA32(a, b, c);
    }
    float bv = ba1[nt * 16 + nloc];
#pragma unroll
    for (int r = 0; r < 8; ++r) {
      float v = c[r] + bv; v = v > 0.f ? v : 0.f;
      H[(mbase + r) * 256 + nt * 16 + nloc] = (_Float16)v;
    }
  }
  WAIT_DS();
  { // tiny head: 16 rows x 2 outputs, one dot per lane
    int r = lane & 15, o = lane >> 4;
    float s = ba2[o];
    for (int k2 = 0; k2 < 256; ++k2) s += (float)H[r * 256 + k2] * wa2_s[k2 * 2 + o];
    u_s[wib * 32 + o * 16 + r] = tanhf(s);   // SCALE == 1
  }
  WAIT_DS();
  if (lane < 16) {
    int node = row0 + lane;
    float u0 = u_s[wib * 32 + lane], u1 = u_s[wib * 32 + 16 + lane];
    float* p  = prev + (size_t)node * 4;
    float* ps = prev_st + (size_t)node * 4;
    float px = p[0] + p[2]*DT_ + 0.5f*u0*DT_*DT_;
    float py = p[1] + p[3]*DT_ + 0.5f*u1*DT_*DT_;
    float vx = p[2] + u0*DT_, vy = p[3] + u1*DT_;
    p[0]=px; p[1]=py; p[2]=vx; p[3]=vy;
    float sx = ps[0] + ps[2]*DT_ + 0.5f*u0*DT_*DT_;
    float sy = ps[1] + ps[3]*DT_ + 0.5f*u1*DT_*DT_;
    float svx = ps[2] + u0*DT_, svy = ps[3] + u1*DT_;
    ps[0]=sx; ps[1]=sy; ps[2]=svx; ps[3]=svy;
    float* os = out_state + ((size_t)t * N_ + node) * 4;
    os[0]=px; os[1]=py; os[2]=vx; os[3]=vy;
    float* oi = out_input + ((size_t)t * N_ + node) * 2;
    oi[0]=u0; oi[1]=u1;
  }
}

__global__ void k_final(const float* __restrict__ acc, float* __restrict__ out) {
  if (threadIdx.x == 0 && blockIdx.x == 0) {
    float t = 0.f;
    for (int i = 0; i < FT_; ++i) t += sqrtf(acc[i]);
    out[0] = t / (float)N_;
    out[1] = acc[12];
  }
}

// ======================================================================
extern "C" void kernel_launch(void* const* d_in, const int* in_sizes, int n_in,
                              void* d_out, int out_size, void* d_ws, size_t ws_size,
                              hipStream_t stream) {
  const float* x_last    = (const float*)d_in[0];
  const float* x_st_last = (const float*)d_in[1];
  const float* hist_enc  = (const float*)d_in[2];
  const float* z         = (const float*)d_in[3];
  const float* node_size = (const float*)d_in[4];
  const float* W_h0 = (const float*)d_in[5];  const float* b_h0 = (const float*)d_in[6];
  const float* W_c0 = (const float*)d_in[7];  const float* b_c0 = (const float*)d_in[8];
  const float* W_gh = (const float*)d_in[9];  const float* b_gh = (const float*)d_in[10];
  const float* W_gru_ih = (const float*)d_in[11];
  const float* W_gru_hh = (const float*)d_in[12];
  const float* b_gru_ih = (const float*)d_in[13];
  const float* b_gru_hh = (const float*)d_in[14];
  const float* W_proj = (const float*)d_in[15]; const float* b_proj = (const float*)d_in[16];
  const float* W_att1 = (const float*)d_in[17]; const float* W_att2 = (const float*)d_in[18];
  const float* v_att  = (const float*)d_in[19];
  const float* W_lstm_ih = (const float*)d_in[20];
  const float* W_lstm_hh = (const float*)d_in[21];
  const float* b_lstm_ih = (const float*)d_in[22];
  const float* b_lstm_hh = (const float*)d_in[23];
  const float* W_e1 = (const float*)d_in[24]; const float* b_e1 = (const float*)d_in[25];
  const float* W_e2 = (const float*)d_in[26]; const float* b_e2 = (const float*)d_in[27];
  const float* W_a1 = (const float*)d_in[28]; const float* b_a1 = (const float*)d_in[29];
  const float* W_a2 = (const float*)d_in[30]; const float* b_a2 = (const float*)d_in[31];
  const int* edge_idx1 = (const int*)d_in[32];
  const int* edge_idx2 = (const int*)d_in[33];
  const int* node_obs_idx = (const int*)d_in[34];

  // ---------------- workspace layout ----------------
  _Float16* hb = (_Float16*)d_ws;
  size_t ho = 0;
  auto aH = [&](size_t n) { _Float16* p = hb + ho; ho += n; return p; };
  _Float16* xzh   = aH((size_t)N_ * 288);
  _Float16* Wgp   = aH((size_t)9 * 64 * 512);     // [ktile, ntile] packed
  _Float16* Whhp  = aH((size_t)8 * 48 * 512);
  _Float16* We1p  = aH((size_t)1 * 16 * 512);
  _Float16* We2p  = aH((size_t)8 * 8 * 512);
  _Float16* Wat1p = aH((size_t)4 * 1 * 512);
  _Float16* Wlp   = aH((size_t)12 * 64 * 512);
  _Float16* Wa1p  = aH((size_t)9 * 16 * 512);
  _Float16* hgh   = aH((size_t)N_ * 256);
  _Float16* feat  = aH((size_t)E_ * 32);
  _Float16* enc_all = aH((size_t)(E_ + 1) * 128);
  _Float16* ctxh  = aH((size_t)N_ * 128);
  _Float16* hbuf0 = aH((size_t)N_ * 256);
  _Float16* hbuf1 = aH((size_t)N_ * 256);
  _Float16* a_in  = aH((size_t)N_ * 288);
  size_t nHalf = ho;

  float* fb = (float*)(hb + ((nHalf + 7) & ~(size_t)7));
  size_t fo = 0;
  auto aF = [&](size_t n) { float* p = fb + fo; fo += n; return p; };
  float* gbuf   = aF((size_t)N_ * 1024);
  float* ghh    = aF((size_t)N_ * 768);
  float* hguide = aF((size_t)N_ * 256);
  float* desb   = aF((size_t)FT_ * N_ * 2);
  float* cbuf   = aF((size_t)N_ * 256);
  float* prevb  = aF((size_t)N_ * 4);
  float* prevst = aF((size_t)N_ * 4);
  float* radb   = aF((size_t)E_);
  float* biasg  = aF(1024);
  float* blb    = aF(1024);
  float* accum  = aF(16);        // [0..11]=terr_sq per step, [12]=ccost

  float* out_state = (float*)d_out;
  float* out_input = out_state + (size_t)FT_ * N_ * 4;
  float* out_scal  = out_state + (size_t)FT_ * N_ * 6;

  auto blocks = [](long n, int bs) { return (unsigned)((n + bs - 1) / bs); };

  // ---------------- once-per-call prep ----------------
  k_zero32<<<blocks((long)nHalf / 2, 256), 256, 0, stream>>>((uint32_t*)hb, (long)nHalf / 2);
  k_zero32<<<1, 32, 0, stream>>>((uint32_t*)accum, 16);

  // pack all weights into WMMA B-fragment tile layout (f16)
  k_pack<<<blocks(9L*16*512, 256), 256, 0, stream>>>(W_gh,      Wgp, 264, 256, 9, 0, 64, 0, 16);
  k_pack<<<blocks(9L*48*512, 256), 256, 0, stream>>>(W_gru_ih,  Wgp, 264, 768, 9, 0, 64, 16, 48);
  k_pack<<<blocks(8L*48*512, 256), 256, 0, stream>>>(W_gru_hh,  Whhp, 256, 768, 8, 0, 48, 0, 48);
  k_pack<<<blocks(1L*16*512, 256), 256, 0, stream>>>(W_e1,      We1p, 8, 256, 1, 0, 16, 0, 16);
  k_pack<<<blocks(8L*8*512, 256), 256, 0, stream>>>(W_e2,       We2p, 256, 128, 8, 0, 8, 0, 8);
  k_pack<<<blocks(4L*1*512, 256), 256, 0, stream>>>(W_att1,     Wat1p, 128, 16, 4, 0, 1, 0, 1);
  k_pack<<<blocks(4L*64*512, 256), 256, 0, stream>>>(W_lstm_ih, Wlp, 128, 1024, 4, 0, 64, 0, 64);
  k_pack<<<blocks(8L*64*512, 256), 256, 0, stream>>>(W_lstm_hh, Wlp, 256, 1024, 8, 4, 64, 0, 64);
  k_pack<<<blocks(9L*16*512, 256), 256, 0, stream>>>(W_a1,      Wa1p, 272, 256, 9, 0, 16, 0, 16);

  k_biasg<<<4, 256, 0, stream>>>(b_gh, b_gru_ih, biasg);
  k_addv<<<4, 256, 0, stream>>>(b_lstm_ih, b_lstm_hh, blb, 1024);
  k_xz<<<blocks((long)N_*288, 256), 256, 0, stream>>>(hist_enc, z, xzh);

  // ---------------- guide GRU pre-pass ----------------
  // gbuf = [h0g | ir | iz | inn]
  k_gemm<<<(512*64)/8, 256, 0, stream>>>(xzh, 288, Wgp, gbuf, 1024, biasg, 512, 64, 9);
  k_ginit<<<blocks((long)N_*256, 256), 256, 0, stream>>>(gbuf, hguide, hgh);
  for (int t = 0; t < FT_; ++t) {
    k_gemm<<<(512*48)/8, 256, 0, stream>>>(hgh, 256, Whhp, ghh, 768, b_gru_hh, 512, 48, 8);
    k_gru<<<blocks((long)N_*256, 256), 256, 0, stream>>>(gbuf, ghh, hguide, hgh);
    k_proj<<<blocks((long)N_*2, 256), 256, 0, stream>>>(hguide, W_proj, b_proj, desb + (size_t)t*N_*2);
  }

  // ---------------- rollout init ----------------
  k_hc0<<<blocks((long)N_*256, 256), 256, 0, stream>>>(x_st_last, W_h0, b_h0, W_c0, b_c0, hbuf0, cbuf);
  k_nodes0<<<blocks(N_, 256), 256, 0, stream>>>(x_last, x_st_last, z, prevb, prevst, a_in);
  k_rad<<<blocks(E_, 256), 256, 0, stream>>>(node_size, edge_idx1, edge_idx2, radb);

  // ---------------- 12-step rollout ----------------
  for (int t = 0; t < FT_; ++t) {
    _Float16* hin  = (t & 1) ? hbuf1 : hbuf0;
    _Float16* hout = (t & 1) ? hbuf0 : hbuf1;
    k_edge<<<blocks(E_, 256), 256, 0, stream>>>(prevb, node_size, edge_idx1, edge_idx2,
                                                radb, feat, accum + 12);
    k_enc<<<(E_/16)/4, 128, 0, stream>>>(feat, We1p, b_e1, We2p, b_e2, enc_all);
    k_att<<<N_/4, 128, 0, stream>>>(t, prevst, x_st_last, desb, enc_all, node_obs_idx,
                                    Wat1p, W_att2, v_att, ctxh, a_in, accum);
    k_lstm<<<(512*16)/8, 256, 0, stream>>>(ctxh, hin, Wlp, blb, cbuf, hout, a_in);
    k_act<<<512/4, 128, 0, stream>>>(t, a_in, Wa1p, b_a1, W_a2, b_a2,
                                     prevb, prevst, out_state, out_input);
  }

  k_final<<<1, 32, 0, stream>>>(accum, out_scal);
  (void)in_sizes; (void)n_in; (void)out_size; (void)ws_size;
}
